// Autoformer_21612275434101
// MI455X (gfx1250) — matrix operand order repaired and checked by reading, not compiled, not run on hardware
//
#include <hip/hip_runtime.h>
#include <hip/hip_bf16.h>
#include <math.h>

// Problem constants (match reference)
#define BD   8
#define LSEQ 2048
#define DM   1024
#define TOPK 7

typedef __attribute__((ext_vector_type(16))) __bf16 v16bf;
typedef __attribute__((ext_vector_type(8)))  __bf16 v8bf;
typedef __attribute__((ext_vector_type(4)))  __bf16 v4bf;
typedef __attribute__((ext_vector_type(8)))  float  v8f;

union FragBF { v16bf v; v8bf h[2]; };

// A-fragment (16x32 bf16, ISA 7.12.2 explicit table):
//   lane = (m = l&15, half = l>>4); elems 0..7 -> K = half*8 + e,
//   elems 8..15 -> K = 16 + half*8 + (e-8).
__device__ __forceinline__ FragBF load_frag_A(const __bf16* rowptr, int kk, int half) {
  FragBF f;
  const __bf16* p = rowptr + kk + half * 8;
  f.h[0] = *(const v8bf*)p;
  f.h[1] = *(const v8bf*)(p + 16);
  return f;
}

// B-fragment (32x16 bf16, same K order as A): lane = (n = l&15, half = l>>4);
//   elems 0..7 -> K = half*8 + e, elems 8..15 -> K = 16 + half*8 + (e-8).
__device__ __forceinline__ FragBF load_frag_B(const __bf16* colptr, int kk, int half) {
  FragBF f;
  const __bf16* p = colptr + kk + half * 8;
  f.h[0] = *(const v8bf*)p;
  f.h[1] = *(const v8bf*)(p + 16);
  return f;
}

// ---------------------------------------------------------------------------
// fp32 -> bf16 conversion (vectorized x4)
// ---------------------------------------------------------------------------
__global__ void f32_to_bf16_kernel(const float* __restrict__ src,
                                   __bf16* __restrict__ dst, int n4) {
  int i = blockIdx.x * blockDim.x + threadIdx.x;
  if (i >= n4) return;
  const float4 f = ((const float4*)src)[i];
  v4bf o;
  o[0] = (__bf16)f.x; o[1] = (__bf16)f.y; o[2] = (__bf16)f.z; o[3] = (__bf16)f.w;
  ((v4bf*)dst)[i] = o;
}

// ---------------------------------------------------------------------------
// W (DM x DM fp32, row-major) -> WT (DM x DM bf16, transposed) via LDS tile
// ---------------------------------------------------------------------------
__global__ void transpose_w_bf16_kernel(const float* __restrict__ W,
                                        __bf16* __restrict__ WT) {
  __shared__ float tile[32][33];
  const int tx = threadIdx.x & 31;
  const int ty = threadIdx.x >> 5;            // 0..7
  const int nt = DM / 32;
  const int bx = blockIdx.x % nt;
  const int by = blockIdx.x / nt;
  const int col0 = bx * 32, row0 = by * 32;
#pragma unroll
  for (int r = ty; r < 32; r += 8)
    tile[r][tx] = W[(size_t)(row0 + r) * DM + col0 + tx];
  __syncthreads();
#pragma unroll
  for (int r = ty; r < 32; r += 8)
    WT[(size_t)(col0 + r) * DM + row0 + tx] = (__bf16)tile[tx][r];
}

__global__ void zero_f32_kernel(float* __restrict__ p, int n) {
  int i = blockIdx.x * blockDim.x + threadIdx.x;
  if (i < n) p[i] = 0.f;
}

// ---------------------------------------------------------------------------
// WMMA GEMM: out[M,N] = A[M,K](bf16) * BT[N,K](bf16)^T + bias[N]
// Workgroup: 256 threads = 8 waves, 128x128 tile; wave = 32x64 (2x4 frags).
// ---------------------------------------------------------------------------
template <bool OUT_F32>
__global__ void wmma_gemm_bias_kernel(const __bf16* __restrict__ A,
                                      const __bf16* __restrict__ BT,
                                      const float* __restrict__ bias,
                                      void* __restrict__ out,
                                      int M, int N, int K) {
  const int lane = threadIdx.x & 31;
  const int wid  = threadIdx.x >> 5;          // 0..7
  const int wm   = wid & 3;                   // 4 waves along M
  const int wn   = wid >> 2;                  // 2 waves along N
  const int half = lane >> 4;                 // lane-half select (ISA layout)
  const int ln   = lane & 15;                 // row/col within 16

  const int rowBase = blockIdx.x * 128 + wm * 32;
  const int colBase = blockIdx.y * 128 + wn * 64;

  const __bf16* aRow0 = A  + (size_t)(rowBase + ln) * K;
  const __bf16* aRow1 = A  + (size_t)(rowBase + 16 + ln) * K;
  const __bf16* bCol0 = BT + (size_t)(colBase + ln) * K;

  v8f acc[2][4] = {};

  for (int kk = 0; kk < K; kk += 32) {
    FragBF a[2], b[4];
    a[0] = load_frag_A(aRow0, kk, half);
    a[1] = load_frag_A(aRow1, kk, half);
    __builtin_prefetch((const void*)(aRow0 + kk + 32), 0, 0);   // next K tile
#pragma unroll
    for (int tn = 0; tn < 4; ++tn)
      b[tn] = load_frag_B(bCol0 + (size_t)tn * 16 * K, kk, half);
#pragma unroll
    for (int tm = 0; tm < 2; ++tm)
#pragma unroll
      for (int tn = 0; tn < 4; ++tn)
        acc[tm][tn] = __builtin_amdgcn_wmma_f32_16x16x32_bf16(
            false, a[tm].v, false, b[tn].v, (short)0, acc[tm][tn], false, false);
  }

#pragma unroll
  for (int tm = 0; tm < 2; ++tm)
#pragma unroll
    for (int tn = 0; tn < 4; ++tn) {
      const int col = colBase + tn * 16 + ln;
      const float bv = bias[col];
#pragma unroll
      for (int r = 0; r < 8; ++r) {
        const int row = rowBase + tm * 16 + r + half * 8;
        const float s = acc[tm][tn][r] + bv;
        if (OUT_F32)
          ((float*)out)[(size_t)row * N + col] = s;
        else
          ((__bf16*)out)[(size_t)row * N + col] = (__bf16)s;
      }
    }
}

// ---------------------------------------------------------------------------
// Gram circular-diagonal reduction:
// mean_value[b,tau] = (1/DM) * sum_{i-j==tau (mod L)} <Q[b,i,:], K[b,j,:]>
// Per workgroup: one 128x128 tile of the LxL Gram matrix via WMMA, reduced
// into a 255-bin LDS diagonal histogram, then global atomics.
// ---------------------------------------------------------------------------
__global__ void gram_diag_kernel(const __bf16* __restrict__ Q,
                                 const __bf16* __restrict__ Kmat,
                                 float* __restrict__ mean_value) {
  __shared__ float diag[256];
  const int tid = threadIdx.x;
  const int b = blockIdx.y;
  const int nt = LSEQ / 128;                  // 16
  const int tileI = blockIdx.x / nt;
  const int tileJ = blockIdx.x % nt;
  const int i0 = tileI * 128, j0 = tileJ * 128;

  diag[tid] = 0.f;
  __syncthreads();

  const int lane = tid & 31;
  const int wid  = tid >> 5;
  const int wm   = wid & 3;
  const int wn   = wid >> 2;
  const int half = lane >> 4;
  const int ln   = lane & 15;

  const __bf16* qRow0 = Q + ((size_t)b * LSEQ + i0 + wm * 32 + ln) * DM;
  const __bf16* qRow1 = qRow0 + (size_t)16 * DM;
  const __bf16* kCol0 = Kmat + ((size_t)b * LSEQ + j0 + wn * 64 + ln) * DM;

  v8f acc[2][4] = {};

  for (int kk = 0; kk < DM; kk += 32) {
    FragBF a[2], bfr[4];
    a[0] = load_frag_A(qRow0, kk, half);
    a[1] = load_frag_A(qRow1, kk, half);
    __builtin_prefetch((const void*)(qRow0 + kk + 32), 0, 0);
#pragma unroll
    for (int tn = 0; tn < 4; ++tn)
      bfr[tn] = load_frag_B(kCol0 + (size_t)tn * 16 * DM, kk, half);
#pragma unroll
    for (int tm = 0; tm < 2; ++tm)
#pragma unroll
      for (int tn = 0; tn < 4; ++tn)
        acc[tm][tn] = __builtin_amdgcn_wmma_f32_16x16x32_bf16(
            false, a[tm].v, false, bfr[tn].v, (short)0, acc[tm][tn], false, false);
  }

  // scatter accumulators into LDS diagonal bins: tau_local = i_local - j_local
#pragma unroll
  for (int tm = 0; tm < 2; ++tm)
#pragma unroll
    for (int tn = 0; tn < 4; ++tn) {
      const int dj = wn * 64 + tn * 16 + ln;
#pragma unroll
      for (int r = 0; r < 8; ++r) {
        const int di = wm * 32 + tm * 16 + r + half * 8;
        atomicAdd(&diag[di - dj + 127], acc[tm][tn][r]);
      }
    }
  __syncthreads();

  if (tid < 255) {
    const int tau = (i0 - j0 + tid - 127) & (LSEQ - 1);
    atomicAdd(&mean_value[b * LSEQ + tau], diag[tid] * (1.0f / (float)DM));
  }
}

// ---------------------------------------------------------------------------
// Batch-mean top-7 + per-batch softmax over the selected lags.
// ---------------------------------------------------------------------------
__global__ void topk_softmax_kernel(const float* __restrict__ mean_value,
                                    int* __restrict__ topk_idx,
                                    float* __restrict__ tmp_corr) {
  __shared__ float s[LSEQ];
  __shared__ float rv[256];
  __shared__ int   ri[256];
  __shared__ int   idxs[TOPK];
  const int tid = threadIdx.x;

  for (int t = tid; t < LSEQ; t += 256) {
    float sum = 0.f;
    for (int b = 0; b < BD; ++b) sum += mean_value[b * LSEQ + t];
    s[t] = sum;
  }
  __syncthreads();

  for (int k = 0; k < TOPK; ++k) {
    float mv = -INFINITY; int mi = 0;
    for (int t = tid; t < LSEQ; t += 256)
      if (s[t] > mv) { mv = s[t]; mi = t; }
    rv[tid] = mv; ri[tid] = mi;
    __syncthreads();
    for (int off = 128; off > 0; off >>= 1) {
      if (tid < off) {
        if (rv[tid + off] > rv[tid] ||
            (rv[tid + off] == rv[tid] && ri[tid + off] < ri[tid])) {
          rv[tid] = rv[tid + off]; ri[tid] = ri[tid + off];
        }
      }
      __syncthreads();
    }
    if (tid == 0) { idxs[k] = ri[0]; s[ri[0]] = -INFINITY; topk_idx[k] = ri[0]; }
    __syncthreads();
  }

  if (tid < BD) {
    float w[TOPK], mx = -INFINITY;
    for (int k = 0; k < TOPK; ++k) {
      w[k] = mean_value[tid * LSEQ + idxs[k]];
      mx = fmaxf(mx, w[k]);
    }
    float den = 0.f;
    for (int k = 0; k < TOPK; ++k) { w[k] = __expf(w[k] - mx); den += w[k]; }
    for (int k = 0; k < TOPK; ++k) tmp_corr[tid * TOPK + k] = w[k] / den;
  }
}

// ---------------------------------------------------------------------------
// 7-tap circularly-shifted weighted gather of V (bf16 in, bf16 out):
// agg[b,l,d] = sum_k w[b,k] * V[b, (l+idx[k]) % L, d]
// ---------------------------------------------------------------------------
__global__ void gather_agg_kernel(const __bf16* __restrict__ Vbf,
                                  const int* __restrict__ topk_idx,
                                  const float* __restrict__ tmp_corr,
                                  __bf16* __restrict__ aggbf) {
  __shared__ int   sh_idx[TOPK];
  __shared__ float sh_w[TOPK];
  const int b = blockIdx.x / LSEQ;
  const int l = blockIdx.x % LSEQ;
  if (threadIdx.x < TOPK) {
    sh_idx[threadIdx.x] = topk_idx[threadIdx.x];
    sh_w[threadIdx.x]   = tmp_corr[b * TOPK + threadIdx.x];
  }
  __syncthreads();
  const int d0 = threadIdx.x * 4;
  float a0 = 0.f, a1 = 0.f, a2 = 0.f, a3 = 0.f;
#pragma unroll
  for (int k = 0; k < TOPK; ++k) {
    const int pos = (l + sh_idx[k]) & (LSEQ - 1);
    const v4bf v = *(const v4bf*)(Vbf + ((size_t)(b * LSEQ + pos) * DM + d0));
    const float w = sh_w[k];
    a0 += w * (float)v[0]; a1 += w * (float)v[1];
    a2 += w * (float)v[2]; a3 += w * (float)v[3];
  }
  v4bf o;
  o[0] = (__bf16)a0; o[1] = (__bf16)a1; o[2] = (__bf16)a2; o[3] = (__bf16)a3;
  *(v4bf*)(aggbf + ((size_t)(b * LSEQ + l) * DM + d0)) = o;
}

// ---------------------------------------------------------------------------
extern "C" void kernel_launch(void* const* d_in, const int* in_sizes, int n_in,
                              void* d_out, int out_size, void* d_ws, size_t ws_size,
                              hipStream_t stream) {
  const float* queries = (const float*)d_in[0];
  const float* keys    = (const float*)d_in[1];
  const float* values  = (const float*)d_in[2];
  const float* Wq = (const float*)d_in[3];
  const float* bq = (const float*)d_in[4];
  const float* Wk = (const float*)d_in[5];
  const float* bk = (const float*)d_in[6];
  const float* Wv = (const float*)d_in[7];
  const float* bv = (const float*)d_in[8];
  const float* Wo = (const float*)d_in[9];
  const float* bo = (const float*)d_in[10];

  char* ws = (char*)d_ws;
  const size_t SZ_BLD = (size_t)BD * LSEQ * DM * sizeof(__bf16);   // 32 MB
  const size_t SZ_W   = (size_t)DM * DM * sizeof(__bf16);          //  2 MB

  __bf16* Xbf  = (__bf16*)(ws + 0 * SZ_BLD);   // reused input staging
  __bf16* Qbf  = (__bf16*)(ws + 1 * SZ_BLD);
  __bf16* Kbf  = (__bf16*)(ws + 2 * SZ_BLD);
  __bf16* Vbf  = (__bf16*)(ws + 3 * SZ_BLD);
  __bf16* Abf  = (__bf16*)(ws + 4 * SZ_BLD);   // aggregated (pre-output GEMM)
  __bf16* WTq  = (__bf16*)(ws + 5 * SZ_BLD);
  __bf16* WTk  = (__bf16*)((char*)WTq + SZ_W);
  __bf16* WTv  = (__bf16*)((char*)WTk + SZ_W);
  __bf16* WTo  = (__bf16*)((char*)WTv + SZ_W);
  float*  meanv    = (float*)((char*)WTo + SZ_W);
  int*    topk_idx = (int*)((char*)meanv + (size_t)BD * LSEQ * sizeof(float));
  float*  tmpc     = (float*)((char*)topk_idx + 256);

  const int M = BD * LSEQ, N = DM, K = DM;
  const int n4 = BD * LSEQ * DM / 4;

  dim3 blk(256);
  dim3 gConv((n4 + 255) / 256);
  dim3 gT((DM / 32) * (DM / 32));
  dim3 gGemm(M / 128, N / 128);
  dim3 gGram((LSEQ / 128) * (LSEQ / 128), BD);

  // Weight transposes -> bf16
  transpose_w_bf16_kernel<<<gT, blk, 0, stream>>>(Wq, WTq);
  transpose_w_bf16_kernel<<<gT, blk, 0, stream>>>(Wk, WTk);
  transpose_w_bf16_kernel<<<gT, blk, 0, stream>>>(Wv, WTv);
  transpose_w_bf16_kernel<<<gT, blk, 0, stream>>>(Wo, WTo);

  // Q projection
  f32_to_bf16_kernel<<<gConv, blk, 0, stream>>>(queries, Xbf, n4);
  wmma_gemm_bias_kernel<false><<<gGemm, blk, 0, stream>>>(Xbf, WTq, bq, Qbf, M, N, K);
  // K projection (Xbf reused; stream order guarantees safety)
  f32_to_bf16_kernel<<<gConv, blk, 0, stream>>>(keys, Xbf, n4);
  wmma_gemm_bias_kernel<false><<<gGemm, blk, 0, stream>>>(Xbf, WTk, bk, Kbf, M, N, K);
  // V projection
  f32_to_bf16_kernel<<<gConv, blk, 0, stream>>>(values, Xbf, n4);
  wmma_gemm_bias_kernel<false><<<gGemm, blk, 0, stream>>>(Xbf, WTv, bv, Vbf, M, N, K);

  // Autocorrelation via Gram circular diagonals
  zero_f32_kernel<<<(BD * LSEQ + 255) / 256, blk, 0, stream>>>(meanv, BD * LSEQ);
  gram_diag_kernel<<<gGram, blk, 0, stream>>>(Qbf, Kbf, meanv);

  // top-k lags + softmax weights
  topk_softmax_kernel<<<1, blk, 0, stream>>>(meanv, topk_idx, tmpc);

  // weighted circular gather of V
  gather_agg_kernel<<<BD * LSEQ, blk, 0, stream>>>(Vbf, topk_idx, tmpc, Abf);

  // output projection -> fp32 d_out
  wmma_gemm_bias_kernel<true><<<gGemm, blk, 0, stream>>>(Abf, WTo, bo, (float*)d_out, M, N, K);
}
